// MultiHeadAttention_22428319219846
// MI455X (gfx1250) — matrix operand order repaired and checked by reading, not compiled, not run on hardware
//
#include <hip/hip_runtime.h>
#include <hip/hip_bf16.h>

// MHA forward for MI455X (gfx1250): bf16 WMMA (v_wmma_f32_16x16x32_bf16),
// f32 accumulate, flash-style online softmax stats, attns written 2x total.

#define S_LEN 2048
#define BATCH 4
#define NHEAD 16
#define DHEAD 64
#define DMODEL 1024
#define BS (BATCH * S_LEN)   // 8192

typedef __bf16 bf16_t;
typedef __attribute__((ext_vector_type(2)))  __bf16 v2bf;
typedef __attribute__((ext_vector_type(4)))  __bf16 v4bf;
typedef __attribute__((ext_vector_type(8)))  __bf16 v8bf;
typedef __attribute__((ext_vector_type(16))) __bf16 v16bf;
typedef __attribute__((ext_vector_type(8)))  float  v8f;

// Native f32 -> bf16 conversion (hardware RNE converter).
static __device__ __forceinline__ bf16_t f2bf(float f) { return (bf16_t)f; }

// Packed pair conversion: prefer the packed hardware op when the builtin exists.
static __device__ __forceinline__ v2bf f2bf2(float lo, float hi) {
#if __has_builtin(__builtin_amdgcn_cvt_pk_bf16_f32)
    return __builtin_amdgcn_cvt_pk_bf16_f32(lo, hi);
#else
    v2bf r;
    r.x = (bf16_t)lo;
    r.y = (bf16_t)hi;
    return r;
#endif
}

static __device__ __forceinline__ v4bf f2bf4(float x, float y, float z, float w) {
    v4bf r;
    *((v2bf*)&r)       = f2bf2(x, y);
    *(((v2bf*)&r) + 1) = f2bf2(z, w);
    return r;
}

// A fragment 16x32 bf16 (ISA 7.12.2): lane<16 -> row=lane, K chunks [0,8)+[16,24);
// lane>=16 -> row=lane-16, K chunks [8,16)+[24,32). p0 points at (row0, k0).
static __device__ __forceinline__ v16bf load_a16(const bf16_t* p0, int ldk, int lane) {
    int m  = lane & 15;
    int kb = (lane >> 4) << 3;           // 0 or 8
    const bf16_t* p = p0 + m * ldk + kb;
    v16bf a;
    *((v8bf*)&a)       = *(const v8bf*)(p);
    *(((v8bf*)&a) + 1) = *(const v8bf*)(p + 16);
    return a;
}

// B fragment 32x16 bf16 from N-major LDS tile (Bt[n][k]): lane<16 -> col=lane,
// K chunks [0,8)+[16,24); lane>=16 -> col=lane-16, K chunks [8,16)+[24,32).
static __device__ __forceinline__ v16bf load_b16(const bf16_t* p0, int ldk, int lane) {
    int n  = lane & 15;
    int kh = (lane >> 4) << 3;           // 0 or 8
    const bf16_t* p = p0 + n * ldk + kh;
    v16bf b;
    *((v8bf*)&b)       = *(const v8bf*)(p);
    *(((v8bf*)&b) + 1) = *(const v8bf*)(p + 16);
    return b;
}

static __device__ __forceinline__ v8f wmma_bf16(v16bf a, v16bf b, v8f c) {
    return __builtin_amdgcn_wmma_f32_16x16x32_bf16(false, a, false, b, (short)0, c,
                                                   false, false);
}

// ---------------------------------------------------------------------------
// Kernel 1: per-head QKV projection. grid = (head 16, mtile 64, which 3).
// Block 256 = 8 waves; block tile 128(M) x 64(N, == one head); K loop 1024/32.
// ---------------------------------------------------------------------------
__global__ __launch_bounds__(256) void qkv_proj_kernel(
    const float* __restrict__ q, const float* __restrict__ k, const float* __restrict__ v,
    const float* __restrict__ wgt_q, const float* __restrict__ wgt_k,
    const float* __restrict__ wgt_v,
    bf16_t* __restrict__ qh, bf16_t* __restrict__ kh, bf16_t* __restrict__ vh)
{
    const int h     = blockIdx.x;
    const int mtile = blockIdx.y;
    const int which = blockIdx.z;
    const float* X = (which == 0) ? q : (which == 1) ? k : v;
    const float* W = (which == 0) ? wgt_q : (which == 1) ? wgt_k : wgt_v;
    bf16_t*      Y = (which == 0) ? qh : (which == 1) ? kh : vh;

    __shared__ __align__(16) bf16_t lds_a[128 * 32];   // [row][k]
    __shared__ __align__(16) bf16_t lds_bt[64 * 32];   // [n][k] (N-major)

    const int t    = threadIdx.x;
    const int lane = t & 31;
    const int wid  = t >> 5;
    const int wm   = wid >> 1;            // 0..3
    const int wn   = wid & 1;             // 0..1
    const int hi   = lane >> 4;
    const int ln   = lane & 15;
    const int rowBase = mtile * 128;

    v8f c[2][2] = {};

    for (int kt = 0; kt < DMODEL; kt += 32) {
        __syncthreads();
        // stage A: 128 rows x 32 k, f32 -> bf16
        {
            const int r0 = t >> 3;
            const int k4 = (t & 7) * 4;
#pragma unroll
            for (int rr = 0; rr < 4; ++rr) {
                const int row = r0 + rr * 32;
                const float4 f = *(const float4*)(X + (size_t)(rowBase + row) * DMODEL + kt + k4);
                *(v4bf*)(&lds_a[row * 32 + k4]) = f2bf4(f.x, f.y, f.z, f.w);
            }
        }
        // stage Bt: Bt[n][kk] = W[h][kt+kk][n] (transpose during staging)
        {
            const int n4 = (t & 15) * 4;
#pragma unroll
            for (int ii = 0; ii < 2; ++ii) {
                const int kk = (t >> 4) + ii * 16;
                const float4 f = *(const float4*)(W + ((size_t)h * DMODEL + kt + kk) * DHEAD + n4);
                lds_bt[(n4 + 0) * 32 + kk] = f2bf(f.x);
                lds_bt[(n4 + 1) * 32 + kk] = f2bf(f.y);
                lds_bt[(n4 + 2) * 32 + kk] = f2bf(f.z);
                lds_bt[(n4 + 3) * 32 + kk] = f2bf(f.w);
            }
        }
        __syncthreads();

        const v16bf a0 = load_a16(&lds_a[(wm * 32) * 32],      32, lane);
        const v16bf a1 = load_a16(&lds_a[(wm * 32 + 16) * 32], 32, lane);
        const v16bf b0 = load_b16(&lds_bt[(wn * 32) * 32],      32, lane);
        const v16bf b1 = load_b16(&lds_bt[(wn * 32 + 16) * 32], 32, lane);
        c[0][0] = wmma_bf16(a0, b0, c[0][0]);
        c[0][1] = wmma_bf16(a0, b1, c[0][1]);
        c[1][0] = wmma_bf16(a1, b0, c[1][0]);
        c[1][1] = wmma_bf16(a1, b1, c[1][1]);
    }

    // store Y[h][row][col] bf16  (Y layout: [H][BS][64])
#pragma unroll
    for (int mi = 0; mi < 2; ++mi)
#pragma unroll
        for (int ni = 0; ni < 2; ++ni)
#pragma unroll
            for (int r = 0; r < 8; ++r) {
                const int row = rowBase + wm * 32 + mi * 16 + r + hi * 8;
                const int col = wn * 32 + ni * 16 + ln;
                Y[(size_t)h * BS * DHEAD + (size_t)row * DHEAD + col] = f2bf(c[mi][ni][r]);
            }
}

// ---------------------------------------------------------------------------
// Kernel 2: scores = scale * Q K^T (WMMA) -> raw scores streamed to attns,
// online softmax stats (row max m, denom l) flash-style. grid=(mtile16, hb64).
// ---------------------------------------------------------------------------
__global__ __launch_bounds__(256) void attn_scores_kernel(
    const bf16_t* __restrict__ qh, const bf16_t* __restrict__ kh,
    float* __restrict__ attns, float* __restrict__ m_buf, float* __restrict__ rl_buf)
{
    const int mtile = blockIdx.x;          // 0..15
    const int hb    = blockIdx.y;          // h*B+b, 0..63
    __shared__ __align__(16) bf16_t lds_q[128 * 64];   // [row][k], resident
    __shared__ __align__(16) bf16_t lds_k[64 * 64];    // [n][k] (kh rows are already N-major)

    const int t    = threadIdx.x;
    const int lane = t & 31;
    const int wid  = t >> 5;
    const int wm   = wid >> 1;
    const int wn   = wid & 1;
    const int hi   = lane >> 4;
    const int ln   = lane & 15;
    const size_t hbase = (size_t)hb * S_LEN * DHEAD;

    // stage Q tile (contiguous 16KB) once
    {
        const uint4* src = (const uint4*)(qh + hbase + (size_t)mtile * 128 * DHEAD);
        uint4* dst = (uint4*)lds_q;
#pragma unroll
        for (int i = 0; i < 4; ++i) dst[t + i * 256] = src[t + i * 256];
    }
    __syncthreads();

    v16bf aq[2][2];
#pragma unroll
    for (int mi = 0; mi < 2; ++mi)
#pragma unroll
        for (int ks = 0; ks < 2; ++ks)
            aq[mi][ks] = load_a16(&lds_q[(wm * 32 + mi * 16) * 64 + ks * 32], 64, lane);

    float mrow[16], lrow[16];
#pragma unroll
    for (int i = 0; i < 16; ++i) { mrow[i] = -1e30f; lrow[i] = 0.0f; }

    const int m0 = mtile * 128 + wm * 32;

    for (int nc = 0; nc < S_LEN / 64; ++nc) {
        __syncthreads();
        {   // stage 64 key rows (contiguous 8KB)
            const uint4* src = (const uint4*)(kh + hbase + (size_t)nc * 64 * DHEAD);
            uint4* dst = (uint4*)lds_k;
            dst[t] = src[t];
            dst[t + 256] = src[t + 256];
            __builtin_prefetch(kh + hbase + (size_t)(nc + 1) * 64 * DHEAD, 0, 1);
        }
        __syncthreads();

        v8f c[2][2] = {};
#pragma unroll
        for (int ks = 0; ks < 2; ++ks) {
            const v16bf b0 = load_b16(&lds_k[(wn * 32) * 64 + ks * 32],      64, lane);
            const v16bf b1 = load_b16(&lds_k[(wn * 32 + 16) * 64 + ks * 32], 64, lane);
            c[0][0] = wmma_bf16(aq[0][ks], b0, c[0][0]);
            c[0][1] = wmma_bf16(aq[0][ks], b1, c[0][1]);
            c[1][0] = wmma_bf16(aq[1][ks], b0, c[1][0]);
            c[1][1] = wmma_bf16(aq[1][ks], b1, c[1][1]);
        }

#pragma unroll
        for (int mi = 0; mi < 2; ++mi)
#pragma unroll
            for (int r = 0; r < 8; ++r) {
                const float v0 = c[mi][0][r] * 0.125f;   // 1/sqrt(64)
                const float v1 = c[mi][1][r] * 0.125f;
                const int qrow = m0 + mi * 16 + r + hi * 8;
                const size_t rowoff = ((size_t)hb * S_LEN + qrow) * S_LEN;
                attns[rowoff + nc * 64 + wn * 32 + ln]      = v0;
                attns[rowoff + nc * 64 + wn * 32 + 16 + ln] = v1;

                float mx = fmaxf(v0, v1);
#pragma unroll
                for (int off = 8; off >= 1; off >>= 1)
                    mx = fmaxf(mx, __shfl_xor(mx, off, 32));
                const int idx = mi * 8 + r;
                const float mo = mrow[idx];
                const float mn = fmaxf(mo, mx);
                float se = __expf(v0 - mn) + __expf(v1 - mn);
#pragma unroll
                for (int off = 8; off >= 1; off >>= 1)
                    se += __shfl_xor(se, off, 32);
                lrow[idx] = lrow[idx] * __expf(mo - mn) + se;
                mrow[idx] = mn;
            }
    }

    if (ln == 0) {
#pragma unroll
        for (int mi = 0; mi < 2; ++mi)
#pragma unroll
            for (int r = 0; r < 8; ++r) {
                const int qrow = m0 + mi * 16 + r + hi * 8;
                m_buf[(size_t)hb * S_LEN + qrow]  = mrow[mi * 8 + r];
                rl_buf[(size_t)hb * S_LEN + qrow] = 1.0f / lrow[mi * 8 + r];
            }
    }
}

// ---------------------------------------------------------------------------
// Kernel 3: normalize attns in place (final probabilities) while feeding the
// same tiles as bf16 WMMA A operands for O = P V.  grid=(mtile16, hb64).
// ---------------------------------------------------------------------------
__global__ __launch_bounds__(256) void attn_pv_kernel(
    float* __restrict__ attns, const bf16_t* __restrict__ vh,
    const float* __restrict__ m_buf, const float* __restrict__ rl_buf,
    bf16_t* __restrict__ ocat)
{
    const int mtile = blockIdx.x;
    const int hb    = blockIdx.y;
    const int h = hb >> 2, b = hb & 3;
    __shared__ __align__(16) bf16_t lds_p[128 * 32];   // [row][k]
    __shared__ __align__(16) bf16_t lds_v[64 * 32];    // [n][k]

    const int t    = threadIdx.x;
    const int lane = t & 31;
    const int wid  = t >> 5;
    const int wm   = wid >> 1;
    const int wn   = wid & 1;
    const int hi   = lane >> 4;
    const int ln   = lane & 15;
    const size_t hbase = (size_t)hb * S_LEN * DHEAD;
    const int rowBase = mtile * 128;

    // per-thread softmax stats for the 4 rows this thread stages (fixed over kt)
    const int r0 = t >> 3;
    const int k4 = (t & 7) * 4;
    float mm4[4], rl4[4];
#pragma unroll
    for (int rr = 0; rr < 4; ++rr) {
        const int qrow = rowBase + r0 + rr * 32;
        mm4[rr] = m_buf[(size_t)hb * S_LEN + qrow];
        rl4[rr] = rl_buf[(size_t)hb * S_LEN + qrow];
    }

    v8f c[2][2] = {};

    for (int kt = 0; kt < S_LEN; kt += 32) {
        __syncthreads();
        // stage P: read raw score, write normalized prob back, bf16 copy to LDS
#pragma unroll
        for (int rr = 0; rr < 4; ++rr) {
            const int row  = r0 + rr * 32;
            const int qrow = rowBase + row;
            const size_t off = ((size_t)hb * S_LEN + qrow) * S_LEN + kt + k4;
            const float4 s = *(const float4*)(attns + off);
            float4 p;
            p.x = __expf(s.x - mm4[rr]) * rl4[rr];
            p.y = __expf(s.y - mm4[rr]) * rl4[rr];
            p.z = __expf(s.z - mm4[rr]) * rl4[rr];
            p.w = __expf(s.w - mm4[rr]) * rl4[rr];
            *(float4*)(attns + off) = p;
            *(v4bf*)(&lds_p[row * 32 + k4]) = f2bf4(p.x, p.y, p.z, p.w);
        }
        // stage V transposed: lds_v[n][kk] = vh[hb][kt+kk][n]
        {
            const int kk = t >> 3;           // 0..31
            const int n8 = (t & 7) * 8;      // 0..56
            const v8bf vv = *(const v8bf*)(vh + hbase + (size_t)(kt + kk) * DHEAD + n8);
#pragma unroll
            for (int j = 0; j < 8; ++j) lds_v[(n8 + j) * 32 + kk] = vv[j];
        }
        __syncthreads();

        const v16bf a0 = load_a16(&lds_p[(wm * 32) * 32],      32, lane);
        const v16bf a1 = load_a16(&lds_p[(wm * 32 + 16) * 32], 32, lane);
        const v16bf b0 = load_b16(&lds_v[(wn * 32) * 32],      32, lane);
        const v16bf b1 = load_b16(&lds_v[(wn * 32 + 16) * 32], 32, lane);
        c[0][0] = wmma_bf16(a0, b0, c[0][0]);
        c[0][1] = wmma_bf16(a0, b1, c[0][1]);
        c[1][0] = wmma_bf16(a1, b0, c[1][0]);
        c[1][1] = wmma_bf16(a1, b1, c[1][1]);
    }

    // O concat layout: ocat[b][s][h*64 + dv], bf16
#pragma unroll
    for (int mi = 0; mi < 2; ++mi)
#pragma unroll
        for (int ni = 0; ni < 2; ++ni)
#pragma unroll
            for (int r = 0; r < 8; ++r) {
                const int srow = rowBase + wm * 32 + mi * 16 + r + hi * 8;
                const int col  = wn * 32 + ni * 16 + ln;
                ocat[((size_t)b * S_LEN + srow) * DMODEL + h * DHEAD + col] =
                    f2bf(c[mi][ni][r]);
            }
}

// ---------------------------------------------------------------------------
// Kernel 4: y = O_cat @ proj_w + proj_b + residual(q).  grid=(ntile16, mtile64).
// ---------------------------------------------------------------------------
__global__ __launch_bounds__(256) void proj_kernel(
    const bf16_t* __restrict__ ocat, const float* __restrict__ pw,
    const float* __restrict__ pb, const float* __restrict__ residual,
    float* __restrict__ y)
{
    const int ntile = blockIdx.x;
    const int mtile = blockIdx.y;
    __shared__ __align__(16) bf16_t lds_a[128 * 32];
    __shared__ __align__(16) bf16_t lds_bt[64 * 32];

    const int t    = threadIdx.x;
    const int lane = t & 31;
    const int wid  = t >> 5;
    const int wm   = wid >> 1;
    const int wn   = wid & 1;
    const int hi   = lane >> 4;
    const int ln   = lane & 15;

    v8f c[2][2] = {};

    for (int kt = 0; kt < DMODEL; kt += 32) {
        __syncthreads();
        {   // stage A (bf16 copy): 128 x 32
            const int row = t >> 1;
            const int off = (t & 1) * 16;
            const uint4* src =
                (const uint4*)(ocat + (size_t)(mtile * 128 + row) * DMODEL + kt + off);
            uint4* dst = (uint4*)(&lds_a[row * 32 + off]);
            dst[0] = src[0];
            dst[1] = src[1];
        }
        {   // stage Bt: Bt[n][kk] = pw[kt+kk][ntile*64+n]
            const int n4 = (t & 15) * 4;
#pragma unroll
            for (int ii = 0; ii < 2; ++ii) {
                const int kk = (t >> 4) + ii * 16;
                const float4 f =
                    *(const float4*)(pw + (size_t)(kt + kk) * DMODEL + ntile * 64 + n4);
                lds_bt[(n4 + 0) * 32 + kk] = f2bf(f.x);
                lds_bt[(n4 + 1) * 32 + kk] = f2bf(f.y);
                lds_bt[(n4 + 2) * 32 + kk] = f2bf(f.z);
                lds_bt[(n4 + 3) * 32 + kk] = f2bf(f.w);
            }
        }
        __syncthreads();

        const v16bf a0 = load_a16(&lds_a[(wm * 32) * 32],      32, lane);
        const v16bf a1 = load_a16(&lds_a[(wm * 32 + 16) * 32], 32, lane);
        const v16bf b0 = load_b16(&lds_bt[(wn * 32) * 32],      32, lane);
        const v16bf b1 = load_b16(&lds_bt[(wn * 32 + 16) * 32], 32, lane);
        c[0][0] = wmma_bf16(a0, b0, c[0][0]);
        c[0][1] = wmma_bf16(a0, b1, c[0][1]);
        c[1][0] = wmma_bf16(a1, b0, c[1][0]);
        c[1][1] = wmma_bf16(a1, b1, c[1][1]);
    }

#pragma unroll
    for (int mi = 0; mi < 2; ++mi)
#pragma unroll
        for (int ni = 0; ni < 2; ++ni)
#pragma unroll
            for (int r = 0; r < 8; ++r) {
                const int row = mtile * 128 + wm * 32 + mi * 16 + r + hi * 8;
                const int col = ntile * 64 + wn * 32 + ni * 16 + ln;
                y[(size_t)row * DMODEL + col] =
                    c[mi][ni][r] + pb[col] + residual[(size_t)row * DMODEL + col];
            }
}

// ---------------------------------------------------------------------------
// Kernel 5: in-place LayerNorm over last dim. One block per row.
// ---------------------------------------------------------------------------
__global__ __launch_bounds__(256) void layernorm_kernel(
    float* __restrict__ y, const float* __restrict__ g, const float* __restrict__ bb)
{
    const int row = blockIdx.x;
    const int t = threadIdx.x;
    const int c0 = t * 4;
    const float4 xv = *(const float4*)(y + (size_t)row * DMODEL + c0);
    float s  = xv.x + xv.y + xv.z + xv.w;
    float s2 = xv.x * xv.x + xv.y * xv.y + xv.z * xv.z + xv.w * xv.w;
#pragma unroll
    for (int off = 16; off >= 1; off >>= 1) {
        s  += __shfl_xor(s,  off, 32);
        s2 += __shfl_xor(s2, off, 32);
    }
    __shared__ float ps[8], ps2[8];
    if ((t & 31) == 0) { ps[t >> 5] = s; ps2[t >> 5] = s2; }
    __syncthreads();
    float ts = 0.0f, ts2 = 0.0f;
#pragma unroll
    for (int i = 0; i < 8; ++i) { ts += ps[i]; ts2 += ps2[i]; }
    const float mu  = ts * (1.0f / DMODEL);
    const float var = ts2 * (1.0f / DMODEL) - mu * mu;
    const float inv = rsqrtf(var + 1e-6f);
    const float4 gv = *(const float4*)(g + c0);
    const float4 bv = *(const float4*)(bb + c0);
    float4 o;
    o.x = (xv.x - mu) * inv * gv.x + bv.x;
    o.y = (xv.y - mu) * inv * gv.y + bv.y;
    o.z = (xv.z - mu) * inv * gv.z + bv.z;
    o.w = (xv.w - mu) * inv * gv.w + bv.w;
    *(float4*)(y + (size_t)row * DMODEL + c0) = o;
}

// ---------------------------------------------------------------------------
extern "C" void kernel_launch(void* const* d_in, const int* in_sizes, int n_in,
                              void* d_out, int out_size, void* d_ws, size_t ws_size,
                              hipStream_t stream) {
    (void)in_sizes; (void)n_in; (void)out_size; (void)ws_size;
    const float* q  = (const float*)d_in[0];
    const float* k  = (const float*)d_in[1];
    const float* v  = (const float*)d_in[2];
    const float* wq = (const float*)d_in[3];
    const float* wk = (const float*)d_in[4];
    const float* wv = (const float*)d_in[5];
    const float* pw = (const float*)d_in[6];
    const float* pb = (const float*)d_in[7];
    const float* lg = (const float*)d_in[8];
    const float* lb = (const float*)d_in[9];

    float* out   = (float*)d_out;                         // [B,S,DMODEL]
    float* attns = out + (size_t)BS * DMODEL;             // [H,B,S,S]

    char* ws = (char*)d_ws;
    bf16_t* qh    = (bf16_t*)(ws);                                   // 16 MB
    bf16_t* kh    = (bf16_t*)(ws + ((size_t)16 << 20));              // 16 MB
    bf16_t* vh    = (bf16_t*)(ws + ((size_t)32 << 20));              // 16 MB
    bf16_t* ocat  = (bf16_t*)(ws + ((size_t)48 << 20));              // 16 MB
    float*  m_buf = (float*)(ws + ((size_t)64 << 20));               // 512 KB
    float*  rlbuf = (float*)(ws + ((size_t)64 << 20) + (512u << 10));// 512 KB

    qkv_proj_kernel<<<dim3(NHEAD, BS / 128, 3), 256, 0, stream>>>(
        q, k, v, wq, wk, wv, qh, kh, vh);
    attn_scores_kernel<<<dim3(S_LEN / 128, NHEAD * BATCH), 256, 0, stream>>>(
        qh, kh, attns, m_buf, rlbuf);
    attn_pv_kernel<<<dim3(S_LEN / 128, NHEAD * BATCH), 256, 0, stream>>>(
        attns, vh, m_buf, rlbuf, ocat);
    proj_kernel<<<dim3(DMODEL / 64, BS / 128), 256, 0, stream>>>(
        ocat, pw, pb, q, out);
    layernorm_kernel<<<BS, 256, 0, stream>>>(out, lg, lb);
}
